// FNO2DEncoder_91027536871896
// MI455X (gfx1250) — matrix operand order repaired and checked by reading, not compiled, not run on hardware
//
#include <hip/hip_runtime.h>
#include <hip/hip_bf16.h>
#include <math.h>

// ---------------------------------------------------------------------------
// FNO2D encoder for MI455X (gfx1250, wave32, WMMA).
// Pointwise 1x1-conv GEMMs: v_wmma_f32_16x16x32_f16. Each wave covers a
// 64(channels) x 64(pixels) output macro-tile, but walks the 4 pixel
// sub-tiles sequentially through the full K loop so only 4 accumulators are
// live at once (no VGPR spills); the activation slab still streams from HBM
// exactly once per wave.
// Spectral branch = truncated-mode separable DFT (only 32x32 corner modes
// survive, so no full FFT needed), f32 VALU, bandwidth-bound.
// ---------------------------------------------------------------------------

typedef __attribute__((ext_vector_type(16))) _Float16 v16h;
typedef __attribute__((ext_vector_type(8)))  float    v8f;

#define NB     4
#define CIN    3
#define HH     256
#define WW     256
#define HP     264
#define WP     264
#define PPIX   (HP * WP)        // 69696 pixels per (batch,channel) plane
#define WIDTHC 64
#define M1M    32
#define NLAY   3

__device__ __forceinline__ float gelu_erf(float x) {
    return 0.5f * x * (1.0f + erff(x * 0.70710678118654752f));
}

// --------------------------- utility kernels -------------------------------

__global__ void k_fill0(float* __restrict__ p, long n) {
    long i = (long)blockIdx.x * blockDim.x + threadIdx.x;
    long stride = (long)gridDim.x * blockDim.x;
    for (; i < n; i += stride) p[i] = 0.0f;
}

// tw[t] = exp(-2*pi*i*t/264), t = 0..263, interleaved (re, im)
__global__ void k_twiddle(float* __restrict__ tw) {
    int t = blockIdx.x * blockDim.x + threadIdx.x;
    if (t < HP) {
        float ang = -6.283185307179586f * (float)t / (float)HP;
        tw[2 * t]     = cosf(ang);
        tw[2 * t + 1] = sinf(ang);
    }
}

// --------------------------- lift network ----------------------------------
// One thread per (b,h,w): [x0,x1,x2,gx,gy] -> 32 (gelu) -> 64, written into
// the padded [B][64][264][264] tensor (pad region pre-zeroed by k_fill0).
__global__ void k_lift(const float* __restrict__ x,
                       const float* __restrict__ w1, const float* __restrict__ b1,
                       const float* __restrict__ w2, const float* __restrict__ b2,
                       float* __restrict__ X) {
    int idx = blockIdx.x * blockDim.x + threadIdx.x;
    if (idx >= NB * HH * WW) return;
    int w_ = idx & (WW - 1);
    int h  = (idx >> 8) & (HH - 1);
    int b  = idx >> 16;

    float c[5];
    const float* xb = x + ((long)b * CIN) * (HH * WW) + (long)h * WW + w_;
    c[0] = xb[0];
    c[1] = xb[HH * WW];
    c[2] = xb[2 * HH * WW];
    c[3] = (float)h  * (1.0f / 255.0f);   // linspace(0,1,256)
    c[4] = (float)w_ * (1.0f / 255.0f);

    float h1[32];
#pragma unroll
    for (int j = 0; j < 32; ++j) {
        float a = b1[j];
#pragma unroll
        for (int i = 0; i < 5; ++i) a += w1[j * 5 + i] * c[i];
        h1[j] = gelu_erf(a);
    }
    long obase = ((long)b * WIDTHC) * PPIX + (long)h * WP + w_;
    for (int o = 0; o < WIDTHC; ++o) {
        float a = b2[o];
#pragma unroll
        for (int j = 0; j < 32; ++j) a += w2[o * 32 + j] * h1[j];
        X[obase + (long)o * PPIX] = a;
    }
}

// --------------------------- WMMA pointwise GEMM ---------------------------
// Y[b][Co][P] = act( W[Co,Ci] * X[b][Ci][P] + bias ) (+ Add)
// Wave macro-tile: MT*16 output channels x 64 pixels. The 4 pixel sub-tiles
// are processed one at a time through the whole (unrolled) K loop, so only
// MT accumulators are live -> no spills. Activation slab read once per wave;
// the small weight matrix is re-read 4x from cache.
// A layout (16x32 f16): lane l, m = l&15, g = l>>4:
//   elem e<8  -> K = 8g + e ; elem e>=8 -> K = 16 + 8g + (e-8)
// B layout (32x16 f16): lane l -> N = l&15 ; elem e -> K same as A
// C/D (16x16 f32): vgpr r, lane l -> M = r + 8*(l>>4), N = l&15
template <int CI, int MT>
__global__ void k_gemm_wmma(const float* __restrict__ Wq, const float* __restrict__ bias,
                            const float* __restrict__ X,  const float* __restrict__ Add,
                            float* __restrict__ Y, int Co, int gelu_on) {
    const int P = PPIX;                    // 69696, divisible by 64
    int lane  = threadIdx.x & 31;
    int wave  = threadIdx.x >> 5;
    int ntile = blockIdx.x * 8 + wave;     // 64-pixel tiles
    if (ntile >= P / 64) return;           // 1089 tiles
    int nbase = ntile * 64;
    int m0    = blockIdx.y * (MT * 16);

    const float* Xb = X + (long)blockIdx.z * CI * P;
    long boffY      = (long)blockIdx.z * Co * P;

    int mrow = lane & 15;
    int g    = lane >> 4;
    int col  = lane & 15;
    int rh   = lane >> 4;

#pragma unroll
    for (int s = 0; s < 4; ++s) {          // pixel sub-tile (16 px)
        v8f acc[MT];
#pragma unroll
        for (int mt = 0; mt < MT; ++mt)
#pragma unroll
            for (int r = 0; r < 8; ++r) acc[mt][r] = 0.0f;

#pragma unroll
        for (int kt = 0; kt < CI / 32; ++kt) {
            const int k0 = kt * 32;
            // prefetch next K-slab of activations (gfx1250 global_prefetch_b8)
            if (kt + 1 < CI / 32)
                __builtin_prefetch(Xb + (long)(k0 + 32 + lane) * P + nbase + s * 16, 0, 1);

            // B tile: 32(K) x 16(N); lane = pixel column, elems = K in A's order
            const float* xrow = Xb + (long)(k0 + 8 * g) * P + nbase + s * 16 + col;
            v16h Bv;
#pragma unroll
            for (int e = 0; e < 8; ++e) Bv[e] = (_Float16)xrow[(long)e * P];
#pragma unroll
            for (int e = 0; e < 8; ++e) Bv[8 + e] = (_Float16)xrow[(long)(16 + e) * P];

#pragma unroll
            for (int mt = 0; mt < MT; ++mt) {
                const float* wrow = Wq + (long)(m0 + mt * 16 + mrow) * CI + k0 + 8 * g;
                v16h A;
#pragma unroll
                for (int e = 0; e < 8; ++e) A[e]     = (_Float16)wrow[e];
#pragma unroll
                for (int e = 0; e < 8; ++e) A[8 + e] = (_Float16)wrow[16 + e];
                acc[mt] = __builtin_amdgcn_wmma_f32_16x16x32_f16(
                    false, A, false, Bv, (short)0, acc[mt], false, false);
            }
        }

        // epilogue for this pixel sub-tile (frees accumulators before next s)
#pragma unroll
        for (int mt = 0; mt < MT; ++mt)
#pragma unroll
            for (int r = 0; r < 8; ++r) {
                int row = m0 + mt * 16 + r + 8 * rh;
                float v = acc[mt][r] + bias[row];
                if (gelu_on) v = gelu_erf(v);
                long oi = boffY + (long)row * P + nbase + s * 16 + col;
                if (Add) v += Add[oi];
                Y[oi] = v;
            }
    }
}

// --------------------------- spectral branch -------------------------------
// F1[b,c,x,ky] = sum_y X[b,c,x,y] * w[(y*ky)%264], ky<32  (rfft along y, trunc)
__global__ void k_dft_y(const float* __restrict__ X, const float* __restrict__ tw,
                        float* __restrict__ F1) {
    __shared__ float s_tw[2 * HP];
    __shared__ float s_row[8][WP];
    int lane = threadIdx.x & 31, wave = threadIdx.x >> 5;
    for (int t = threadIdx.x; t < 2 * HP; t += 256) s_tw[t] = tw[t];
    long total = (long)NB * WIDTHC * HP;
    long row = (long)blockIdx.x * 8 + wave;    // (b*64+c)*264 + x
    if (row < total) {
        const float* src = X + row * WP;
        for (int y = lane; y < WP; y += 32) s_row[wave][y] = src[y];
    }
    __syncthreads();
    if (row >= total) return;
    int ky = lane;                             // 0..31
    float ar = 0.0f, ai = 0.0f;
    for (int y = 0; y < WP; ++y) {
        int t = (y * ky) % HP;
        float v = s_row[wave][y];
        ar += v * s_tw[2 * t];
        ai += v * s_tw[2 * t + 1];
    }
    float* dst = F1 + (row * 32 + ky) * 2;
    dst[0] = ar; dst[1] = ai;
}

// F2[b,c,kxi,ky] = sum_x F1[b,c,x,ky] * w[(x*kx)%264], kx = kxi<32 ? kxi : kxi+200
__global__ void k_dft_x(const float* __restrict__ F1, const float* __restrict__ tw,
                        float* __restrict__ F2) {
    int idx = blockIdx.x * 256 + threadIdx.x;  // bc*2048 + kxi*32 + ky
    if (idx >= NB * WIDTHC * 64 * 32) return;
    int ky  = idx & 31;
    int kxi = (idx >> 5) & 63;
    int bc  = idx >> 11;
    int kx  = (kxi < 32) ? kxi : kxi + 200;    // rows 232..263 = negative freqs
    const float* base = F1 + ((long)bc * HP * 32 + ky) * 2;
    float ar = 0.0f, ai = 0.0f;
    for (int x = 0; x < HP; ++x) {
        int t = (x * kx) % HP;
        float xr = base[x * 64], xi = base[x * 64 + 1];
        float cr = tw[2 * t],    ci = tw[2 * t + 1];
        ar += xr * cr - xi * ci;
        ai += xr * ci + xi * cr;
    }
    F2[(long)idx * 2]     = ar;
    F2[(long)idx * 2 + 1] = ai;
}

// F3[b,o,kxi,ky] = sum_i F2[b,i,kxi,ky] * (wr + i*wi)[i,o,kx,ky]
__global__ void k_mix(const float* __restrict__ F2,
                      const float* __restrict__ w1r, const float* __restrict__ w1i,
                      const float* __restrict__ w2r, const float* __restrict__ w2i,
                      float* __restrict__ F3) {
    int idx = blockIdx.x * 256 + threadIdx.x;  // b*131072 + o*2048 + kxi*32 + ky
    if (idx >= NB * WIDTHC * 64 * 32) return;
    int ky  = idx & 31;
    int kxi = (idx >> 5) & 63;
    int o   = (idx >> 11) & 63;
    int b   = idx >> 17;
    const float* wr; const float* wi; int kxw;
    if (kxi < 32) { wr = w1r; wi = w1i; kxw = kxi; }
    else          { wr = w2r; wi = w2i; kxw = kxi - 32; }
    long wbase = ((long)o * 32 + kxw) * 32 + ky;            // + i*65536
    const float* xb = F2 + ((long)b * WIDTHC * 2048 + kxi * 32 + ky) * 2; // + i*4096
    float ar = 0.0f, ai = 0.0f;
    for (int i = 0; i < WIDTHC; ++i) {
        float xr = xb[(long)i * 4096], xi = xb[(long)i * 4096 + 1];
        float cr = wr[wbase + (long)i * 65536];
        float ci = wi[wbase + (long)i * 65536];
        ar += xr * cr - xi * ci;
        ai += xr * ci + xi * cr;
    }
    F3[(long)idx * 2]     = ar;
    F3[(long)idx * 2 + 1] = ai;
}

// I1[b,o,x,ky] = sum_kxi F3[b,o,kxi,ky] * conj(w[(x*kx)%264])   (ifft along x)
__global__ void k_idft_x(const float* __restrict__ F3, const float* __restrict__ tw,
                         float* __restrict__ I1) {
    int idx = blockIdx.x * 256 + threadIdx.x;  // (b*64+o)*264*32 + x*32 + ky
    if (idx >= NB * WIDTHC * HP * 32) return;
    int ky = idx & 31;
    int x  = (idx >> 5) % HP;
    int bo = idx / (HP * 32);
    const float* base = F3 + ((long)bo * 2048 + ky) * 2;    // + kxi*64
    float ar = 0.0f, ai = 0.0f;
#pragma unroll 4
    for (int kxi = 0; kxi < 64; ++kxi) {
        int kx = (kxi < 32) ? kxi : kxi + 200;
        int t  = (x * kx) % HP;
        float cr =  tw[2 * t];
        float ci = -tw[2 * t + 1];                          // conjugate -> e^{+i}
        float xr = base[kxi * 64], xi = base[kxi * 64 + 1];
        ar += xr * cr - xi * ci;
        ai += xr * ci + xi * cr;
    }
    I1[(long)idx * 2]     = ar;
    I1[(long)idx * 2 + 1] = ai;
}

// S[b,o,x,y] = (F0.re + 2*sum_{ky=1..31} Re(F[ky] e^{+i 2pi y ky/264})) / 264^2
// Re(F e^{+i th}) = Fr*wr[t] + Fi*wi[t]  (since wi = -sin th)
__global__ void k_idft_y(const float* __restrict__ I1, const float* __restrict__ tw,
                         float* __restrict__ S) {
    __shared__ float s_tw[2 * HP];
    __shared__ float s_f[8][64];
    int lane = threadIdx.x & 31, wave = threadIdx.x >> 5;
    for (int t = threadIdx.x; t < 2 * HP; t += 256) s_tw[t] = tw[t];
    long total = (long)NB * WIDTHC * HP;
    long row = (long)blockIdx.x * 8 + wave;    // (b*64+o)*264 + x
    if (row < total) {
        const float* src = I1 + row * 64;
        s_f[wave][lane]      = src[lane];
        s_f[wave][lane + 32] = src[lane + 32];
    }
    __syncthreads();
    if (row >= total) return;
    float* dst = S + row * WP;
    const float inv = 1.0f / ((float)HP * (float)WP);
    for (int y = lane; y < WP; y += 32) {
        float acc = s_f[wave][0];              // DC: real part
#pragma unroll 4
        for (int ky = 1; ky < 32; ++ky) {
            int t = (y * ky) % HP;
            acc += 2.0f * (s_f[wave][2 * ky] * s_tw[2 * t] +
                           s_f[wave][2 * ky + 1] * s_tw[2 * t + 1]);
        }
        dst[y] = acc * inv;
    }
}

// --------------------------- crop ------------------------------------------
__global__ void k_crop(const float* __restrict__ X, float* __restrict__ out) {
    long idx = (long)blockIdx.x * 256 + threadIdx.x;
    if (idx >= (long)NB * WIDTHC * HH * WW) return;
    int w  = idx & (WW - 1);
    int h  = (idx >> 8) & (HH - 1);
    long bc = idx >> 16;
    out[idx] = X[(bc * HP + h) * WP + w];
}

// --------------------------- driver ----------------------------------------
extern "C" void kernel_launch(void* const* d_in, const int* in_sizes, int n_in,
                              void* d_out, int out_size, void* d_ws, size_t ws_size,
                              hipStream_t stream) {
    (void)in_sizes; (void)n_in; (void)out_size; (void)ws_size;

    const float* x       = (const float*)d_in[0];
    const float* lift_w1 = (const float*)d_in[1];
    const float* lift_b1 = (const float*)d_in[2];
    const float* lift_w2 = (const float*)d_in[3];
    const float* lift_b2 = (const float*)d_in[4];
    const float* conv_w  = (const float*)d_in[5];
    const float* conv_b  = (const float*)d_in[6];
    const float* mlp_w1  = (const float*)d_in[7];
    const float* mlp_b1  = (const float*)d_in[8];
    const float* mlp_w2  = (const float*)d_in[9];
    const float* mlp_b2  = (const float*)d_in[10];
    const float* sp_w1r  = (const float*)d_in[11];
    const float* sp_w1i  = (const float*)d_in[12];
    const float* sp_w2r  = (const float*)d_in[13];
    const float* sp_w2i  = (const float*)d_in[14];
    float* out = (float*)d_out;

    // workspace layout (floats)
    const size_t NX  = (size_t)NB * WIDTHC * PPIX;   // 17,842,176
    float* ws  = (float*)d_ws;
    float* Xa  = ws;
    float* Xb  = Xa + NX;
    float* T2  = Xb + NX;                            // NB*128*P, time-aliased w/ spectral bufs
    float* T1  = T2 + (size_t)NB * 128 * PPIX;
    float* tw  = T1 + NX;
    // spectral scratch lives inside T2's region (disjoint in time)
    float* F1b = T2;                                 // NB*64*264*32*2 = 4,325,376 floats
    float* F2b = F1b + (size_t)NB * WIDTHC * HP * 32 * 2;
    float* F3b = F2b + (size_t)NB * WIDTHC * 64 * 32 * 2;
    float* I1b = F3b + (size_t)NB * WIDTHC * 64 * 32 * 2;

    // init
    k_fill0  <<<2048, 256, 0, stream>>>(Xa, (long)NX);
    k_twiddle<<<(HP + 63) / 64, 64, 0, stream>>>(tw);
    k_lift   <<<(NB * HH * WW + 255) / 256, 256, 0, stream>>>(
        x, lift_w1, lift_b1, lift_w2, lift_b2, Xa);

    const int rows8   = (NB * WIDTHC * HP + 7) / 8;          // 8448 blocks (wave/row)
    const int modeBlk = (NB * WIDTHC * 64 * 32 + 255) / 256; // 2048
    const int ixBlk   = (NB * WIDTHC * HP * 32 + 255) / 256; // 8448
    const int nblkN   = (PPIX / 64 + 7) / 8;                 // 137

    float* cur = Xa;
    float* nxt = Xb;
    for (int k = 0; k < NLAY; ++k) {
        const long spOff = (long)k * WIDTHC * WIDTHC * M1M * M1M;
        // spectral branch: cur -> nxt (residual term)
        k_dft_y <<<rows8, 256, 0, stream>>>(cur, tw, F1b);
        k_dft_x <<<modeBlk, 256, 0, stream>>>(F1b, tw, F2b);
        k_mix   <<<modeBlk, 256, 0, stream>>>(F2b, sp_w1r + spOff, sp_w1i + spOff,
                                              sp_w2r + spOff, sp_w2i + spOff, F3b);
        k_idft_x<<<ixBlk, 256, 0, stream>>>(F3b, tw, I1b);
        k_idft_y<<<rows8, 256, 0, stream>>>(I1b, tw, nxt);
        // pointwise branch (WMMA GEMMs): conv -> mlp1(+gelu) -> mlp2 (+spectral add)
        // 64 channels/wave => activations are streamed from HBM once (conv,
        // mlp2) or twice (mlp1, Co=128) instead of once per 16-channel tile.
        k_gemm_wmma<64, 4><<<dim3(nblkN, 1, NB), 256, 0, stream>>>(
            conv_w + (long)k * 64 * 64, conv_b + (long)k * 64, cur, nullptr, T1, 64, 0);
        k_gemm_wmma<64, 4><<<dim3(nblkN, 2, NB), 256, 0, stream>>>(
            mlp_w1 + (long)k * 128 * 64, mlp_b1 + (long)k * 128, T1, nullptr, T2, 128, 1);
        k_gemm_wmma<128, 4><<<dim3(nblkN, 1, NB), 256, 0, stream>>>(
            mlp_w2 + (long)k * 64 * 128, mlp_b2 + (long)k * 64, T2, nxt, nxt, 64, 0);
        float* t = cur; cur = nxt; nxt = t;
    }

    k_crop<<<(NB * WIDTHC * HH * WW + 255) / 256, 256, 0, stream>>>(cur, out);
}
